// SimpleDeltaNetLayer_21758304321838
// MI455X (gfx1250) — matrix operand order repaired and checked by reading, not compiled, not run on hardware
//
#include <hip/hip_runtime.h>
#include <hip/hip_bf16.h>
#include <math.h>

// ---------------------------------------------------------------------------
// SimpleDeltaNetLayer: B=2, L=2048, D_MODEL=512, H=8, D=64
// Chunked WY-form delta rule, chunk C=64.  bf16 WMMA everywhere; async
// global->LDS staging (ASYNCcnt) for pure-copy tiles.
// ---------------------------------------------------------------------------
#define BB      2
#define LL      2048
#define DMODEL  512
#define HH      8
#define DD      64
#define MROWS   (BB * LL)        // 4096 GEMM rows
#define HD      (HH * DD)        // 512
#define CC      64               // chunk length
#define NC      (LL / CC)        // 32 chunks
#define BH      (BB * HH)        // 16 heads total

typedef __attribute__((ext_vector_type(16))) __bf16 v16bf_t;
typedef __attribute__((ext_vector_type(8)))  float  v8f_t;

#define WMMA_BF16(a, b, c) \
    __builtin_amdgcn_wmma_f32_16x16x32_bf16(false, (a), false, (b), (short)0, (c), false, false)

// ---------------------------------------------------------------------------
// helpers
// ---------------------------------------------------------------------------
static __device__ __forceinline__ unsigned short f2bf(float f) {
    unsigned int u = __float_as_uint(f);
    u += 0x7FFFu + ((u >> 16) & 1u);
    return (unsigned short)(u >> 16);
}
static __device__ __forceinline__ float bf2f(unsigned short s) {
    return __uint_as_float(((unsigned int)s) << 16);
}

// async 16B/lane copy: global -> LDS, tracked by ASYNCcnt
static __device__ __forceinline__ void async_b128_to_lds(const void* gptr, void* lds_generic) {
    unsigned long long ga = (unsigned long long)(size_t)gptr;
    unsigned int       la = (unsigned int)(size_t)lds_generic;
    asm volatile("global_load_async_to_lds_b128 %0, %1, off"
                 :: "v"(la), "v"(ga) : "memory");
}
static __device__ __forceinline__ void async_wait0() {
    asm volatile("s_wait_asynccnt 0x0" ::: "memory");
}

// A-operand fragment (16-bit): lane(m,g) holds K = [kb+8g .. +7] and [kb+16+8g .. +7]
static __device__ __forceinline__ v16bf_t frag_a_ld(const unsigned short* s, int row,
                                                    int stride, int kb, int g) {
    union { v16bf_t v; uint4 q[2]; } u;
    u.q[0] = *reinterpret_cast<const uint4*>(s + row * stride + kb + 8 * g);
    u.q[1] = *reinterpret_cast<const uint4*>(s + row * stride + kb + 16 + 8 * g);
    return u.v;
}
// B-operand fragment: for C = X @ Y, pass Y stored transposed ([n][k] row-major);
// equivalently for C = X @ Y^T pass Y row-major directly. K = [kb+8g .. +7] and [kb+16+8g .. +7].
static __device__ __forceinline__ v16bf_t frag_b_ld(const unsigned short* s, int row,
                                                    int stride, int kb, int g) {
    union { v16bf_t v; uint4 q[2]; } u;
    u.q[0] = *reinterpret_cast<const uint4*>(s + row * stride + kb + 8 * g);
    u.q[1] = *reinterpret_cast<const uint4*>(s + row * stride + kb + 16 + 8 * g);
    return u.v;
}

// ---------------------------------------------------------------------------
// fp32 -> bf16 conversions
// ---------------------------------------------------------------------------
__global__ void cvt_f32_bf16_kernel(const float* __restrict__ in,
                                    unsigned short* __restrict__ out, int n) {
    int i = blockIdx.x * blockDim.x + threadIdx.x;
    if (i < n) out[i] = f2bf(in[i]);
}

// o (B,H,L,D) -> ob (B*L, H*D) bf16
__global__ void o_to_bf16_kernel(const float* __restrict__ o,
                                 unsigned short* __restrict__ ob, int n) {
    int i = blockIdx.x * blockDim.x + threadIdx.x;
    if (i >= n) return;
    int col = i & (HD - 1);
    int bl  = i >> 9;
    int h = col >> 6, d = col & 63;
    int b = bl >> 11, l = bl & (LL - 1);
    size_t src = (((size_t)(b * HH + h) * LL + l) << 6) + d;
    ob[i] = f2bf(o[src]);
}

// ---------------------------------------------------------------------------
// WMMA bf16 GEMM:  C(4096x512) = A(4096x512) * W(512x512)
//   mode 0: plain fp32 row-major store;  mode 1: SiLU + (B,H,L,D) scatter
// A tile staged via GLOBAL_LOAD_ASYNC_TO_LDS_B128; B tile transposed through
// VGPRs (needs the in-flight transpose).
// ---------------------------------------------------------------------------
#define TM 128
#define TN 64
#define TK 32

__global__ __launch_bounds__(256)
void gemm_bf16_kernel(const unsigned short* __restrict__ A,
                      const unsigned short* __restrict__ W,
                      float* __restrict__ out, int mode) {
    __shared__ __align__(16) unsigned short As[TM * TK];  // [m][k]
    __shared__ __align__(16) unsigned short Bs[TN * TK];  // [n][k] (transposed)

    const int tid  = threadIdx.x;
    const int lane = tid & 31;
    const int wave = tid >> 5;
    const int wm   = wave & 3;
    const int wn   = wave >> 2;
    const int lm   = lane & 15;
    const int g    = lane >> 4;

    const int n0 = blockIdx.x * TN;
    const int m0 = blockIdx.y * TM;

    v8f_t acc00 = {}, acc01 = {}, acc10 = {}, acc11 = {};

    for (int kb = 0; kb < DMODEL; kb += TK) {
        // --- A tile: async copy, 2 x 16B per thread, LDS offset == tid*16 ---
        {
            int r0 = tid >> 2, kc0 = tid & 3;
            async_b128_to_lds(A + (size_t)(m0 + r0) * DMODEL + kb + kc0 * 8,
                              &As[r0 * TK + kc0 * 8]);
            int c1 = tid + 256;
            int r1 = c1 >> 2, kc1 = c1 & 3;
            async_b128_to_lds(A + (size_t)(m0 + r1) * DMODEL + kb + kc1 * 8,
                              &As[r1 * TK + kc1 * 8]);
        }
        // --- B tile: load 16B, scatter transposed ---
        {
            int k  = tid >> 3;
            int nc = tid & 7;
            uint4 wv = *reinterpret_cast<const uint4*>(W + (size_t)(kb + k) * HD + n0 + nc * 8);
            const unsigned short* ws = reinterpret_cast<const unsigned short*>(&wv);
            #pragma unroll
            for (int j = 0; j < 8; ++j)
                Bs[(nc * 8 + j) * TK + k] = ws[j];
        }
        async_wait0();
        __syncthreads();

        v16bf_t a0 = frag_a_ld(As, wm * 32 + lm,      TK, 0, g);
        v16bf_t a1 = frag_a_ld(As, wm * 32 + 16 + lm, TK, 0, g);
        v16bf_t b0 = frag_b_ld(Bs, wn * 32 + lm,      TK, 0, g);
        v16bf_t b1 = frag_b_ld(Bs, wn * 32 + 16 + lm, TK, 0, g);

        acc00 = WMMA_BF16(a0, b0, acc00);
        acc01 = WMMA_BF16(a0, b1, acc01);
        acc10 = WMMA_BF16(a1, b0, acc10);
        acc11 = WMMA_BF16(a1, b1, acc11);

        __syncthreads();
    }

    v8f_t accs[4] = { acc00, acc01, acc10, acc11 };
    #pragma unroll
    for (int t = 0; t < 4; ++t) {
        int ti = t >> 1, tj = t & 1;
        #pragma unroll
        for (int r = 0; r < 8; ++r) {
            int row = m0 + wm * 32 + ti * 16 + r + 8 * g;
            int col = n0 + wn * 32 + tj * 16 + lm;
            float val = accs[t][r];
            if (mode == 1) {
                val = val / (1.0f + __expf(-val));          // SiLU
                int b = row >> 11, l = row & (LL - 1);
                int h = col >> 6,  d = col & 63;
                out[(((size_t)(b * HH + h) * LL + l) << 6) + d] = val;
            } else {
                out[(size_t)row * HD + col] = val;
            }
        }
    }
}

// ---------------------------------------------------------------------------
// beta[b,h,l] = sigmoid( x[b,l,:] . Wb[:,h] )
// ---------------------------------------------------------------------------
__global__ void beta_kernel(const float* __restrict__ x, const float* __restrict__ Wb,
                            float* __restrict__ beta) {
    int i = blockIdx.x * blockDim.x + threadIdx.x;
    if (i >= BB * LL * HH) return;
    int h  = i & (HH - 1);
    int bl = i >> 3;
    const float* xr = x + (size_t)bl * DMODEL;
    float s = 0.f;
    #pragma unroll 8
    for (int c = 0; c < DMODEL; ++c) s += xr[c] * Wb[c * HH + h];
    int b = bl >> 11, l = bl & (LL - 1);
    beta[(size_t)(b * HH + h) * LL + l] = 1.0f / (1.0f + __expf(-s));
}

// ---------------------------------------------------------------------------
// Phase A (parallel over bh x chunk = 512 blocks, 256 thr):
//   A  = stril(diag(beta) K K^T)           (WMMA, bf16)
//   P  = tril(Qs K^T)  -> ws (bf16)        (WMMA; Qs = q * D^-0.5)
//   W  = (I+A)^-1 diag(beta) K -> ws bf16  (per-column forward substitution)
//   Ub = (I+A)^-1 diag(beta) V -> ws fp32
//   K^T (bf16), Qs (bf16) -> ws
// ---------------------------------------------------------------------------
__global__ __launch_bounds__(256)
void delta_prep_kernel(const float* __restrict__ qbuf, const float* __restrict__ kbuf,
                       const float* __restrict__ vbuf, const float* __restrict__ bet,
                       unsigned short* __restrict__ Wg,  unsigned short* __restrict__ Qg,
                       unsigned short* __restrict__ Pg,  unsigned short* __restrict__ Ktg,
                       float* __restrict__ Ubg) {
    const int bh = blockIdx.x >> 5;            // /NC
    const int c  = blockIdx.x & (NC - 1);
    const size_t base = ((size_t)bh * LL + c * CC) * DD;   // into q/k/v (B,H,L,D)
    const size_t cb   = ((size_t)bh * NC + c) * CC * DD;   // chunk-major ws

    __shared__ __align__(16) unsigned short Kb[CC * DD];   // [t][dk]   8KB
    __shared__ __align__(16) unsigned short Qb[CC * DD];   // [t][dk]   8KB
    __shared__ unsigned short Amat[CC * CC];               // bf16 A    8KB
    __shared__ float bsh[CC];
    __shared__ float Xs[128 * 65];                         // subst. columns, padded

    const int tid  = threadIdx.x;
    const int lane = tid & 31;
    const int wv   = tid >> 5;
    const int lm   = lane & 15;
    const int g    = lane >> 4;

    // stage + emit K^T / Qs
    for (int i = tid; i < CC * DD; i += 256) {
        int t = i >> 6, d = i & 63;
        float kvv = kbuf[base + i];
        float qv  = qbuf[base + i] * 0.125f;               // D^-0.5
        unsigned short k16 = f2bf(kvv), q16 = f2bf(qv);
        Kb[i] = k16;
        Qb[i] = q16;
        Ktg[cb + (size_t)d * CC + t] = k16;                // transposed
        Qg[cb + i] = q16;
    }
    if (tid < CC) bsh[tid] = bet[(size_t)bh * LL + c * CC + tid];
    __syncthreads();

    // K K^T and Qs K^T  (B operand = rows of K, since B = K^T)
    #pragma unroll
    for (int t2 = 0; t2 < 2; ++t2) {
        int tile = wv + t2 * 8;
        int tm = tile >> 2, tn = tile & 3;
        v8f_t accK = {}, accP = {};
        #pragma unroll
        for (int kb = 0; kb < DD; kb += TK) {
            v16bf_t bfrag = frag_b_ld(Kb, tn * 16 + lm, DD, kb, g);
            v16bf_t ak    = frag_a_ld(Kb, tm * 16 + lm, DD, kb, g);
            v16bf_t aq    = frag_a_ld(Qb, tm * 16 + lm, DD, kb, g);
            accK = WMMA_BF16(ak, bfrag, accK);
            accP = WMMA_BF16(aq, bfrag, accP);
        }
        #pragma unroll
        for (int r = 0; r < 8; ++r) {
            int row = tm * 16 + r + 8 * g;                 // t
            int col = tn * 16 + lm;                        // s
            float av = (col < row) ? bsh[row] * accK[r] : 0.f;
            Amat[row * CC + col] = f2bf(av);
            float pv = (col <= row) ? accP[r] : 0.f;
            Pg[cb + (size_t)row * CC + col] = f2bf(pv);
        }
    }
    __syncthreads();

    // forward substitution: (I+A) X = diag(beta) * [K | V], column per thread
    if (tid < 128) {
        int which = tid >> 6, col = tid & 63;
        const float* src = which ? vbuf : kbuf;
        float* X = &Xs[tid * 65];
        for (int t = 0; t < CC; ++t)
            X[t] = bsh[t] * src[base + (size_t)t * DD + col];
        for (int t = 1; t < CC; ++t) {
            float acc = 0.f;
            for (int s = 0; s < t; ++s)
                acc += bf2f(Amat[t * CC + s]) * X[s];
            X[t] -= acc;
        }
        if (which == 0) {
            for (int t = 0; t < CC; ++t)
                Wg[cb + (size_t)t * DD + col] = f2bf(X[t]);
        } else {
            for (int t = 0; t < CC; ++t)
                Ubg[cb + (size_t)t * DD + col] = X[t];
        }
    }
}

// ---------------------------------------------------------------------------
// Phase B (sequential over chunks; one block per bh, 8 waves, all WMMA):
//   U = Ub - W @ S0 ; O = Qs @ S0 + P @ U ; S = S0 + K^T @ U
// S fp32 in LDS; bf16 transposed copies (St, Ut) feed B operands.
// K^T chunk staged with async global->LDS.
// ---------------------------------------------------------------------------
__global__ __launch_bounds__(256)
void delta_seq_kernel(const unsigned short* __restrict__ Wg,  const unsigned short* __restrict__ Qg,
                      const unsigned short* __restrict__ Pg,  const unsigned short* __restrict__ Ktg,
                      const float* __restrict__ Ubg, float* __restrict__ obuf) {
    const int bh = blockIdx.x;
    __shared__ float Sf[DD * DD];                          // S[dk][dv] fp32 16KB
    __shared__ __align__(16) unsigned short St[DD * DD];   // S^T bf16 [dv][dk]
    __shared__ __align__(16) unsigned short Ut[CC * DD];   // U^T bf16 [dv][t]
    __shared__ __align__(16) unsigned short Kt[DD * CC];   // K^T bf16 [dk][t]

    const int tid  = threadIdx.x;
    const int lane = tid & 31;
    const int wv   = tid >> 5;
    const int lm   = lane & 15;
    const int g    = lane >> 4;

    for (int i = tid; i < DD * DD; i += 256) { Sf[i] = 0.f; St[i] = 0; }
    __syncthreads();

    for (int c = 0; c < NC; ++c) {
        const size_t cb = ((size_t)bh * NC + c) * CC * DD;

        // stage K^T chunk (8KB) via async global->LDS: 2 x 16B per thread
        {
            const unsigned short* src = Ktg + cb;
            async_b128_to_lds(src + tid * 8,          &Kt[tid * 8]);
            async_b128_to_lds(src + tid * 8 + 2048,   &Kt[tid * 8 + 2048]);
        }
        // prefetch next chunk's operands (global_prefetch_b8)
        if (c + 1 < NC) {
            size_t nb = cb + (size_t)CC * DD;
            __builtin_prefetch(Wg  + nb, 0, 1);
            __builtin_prefetch(Qg  + nb, 0, 1);
            __builtin_prefetch(Pg  + nb, 0, 1);
            __builtin_prefetch(Ktg + nb, 0, 1);
            __builtin_prefetch(Ubg + nb, 0, 1);
        }
        async_wait0();
        __syncthreads();

        // ---- M1: U = Ub - W @ S0 ; write U^T bf16
        #pragma unroll
        for (int t2 = 0; t2 < 2; ++t2) {
            int tile = wv + t2 * 8;
            int tm = tile >> 2, tn = tile & 3;
            v8f_t acc = {};
            #pragma unroll
            for (int kb = 0; kb < DD; kb += TK) {
                v16bf_t a = frag_a_ld(Wg + cb, tm * 16 + lm, DD, kb, g);
                v16bf_t b = frag_b_ld(St,      tn * 16 + lm, DD, kb, g);
                acc = WMMA_BF16(a, b, acc);
            }
            #pragma unroll
            for (int r = 0; r < 8; ++r) {
                int row = tm * 16 + r + 8 * g;             // t
                int col = tn * 16 + lm;                    // dv
                float u = Ubg[cb + (size_t)row * DD + col] - acc[r];
                Ut[col * CC + row] = f2bf(u);
            }
        }
        __syncthreads();

        // ---- M2: O = Qs @ S0 + P @ U   ;  ---- M3: S += K^T @ U
        #pragma unroll
        for (int t2 = 0; t2 < 2; ++t2) {
            int tile = wv + t2 * 8;
            int tm = tile >> 2, tn = tile & 3;

            v8f_t accO = {};
            #pragma unroll
            for (int kb = 0; kb < DD; kb += TK) {
                v16bf_t a = frag_a_ld(Qg + cb, tm * 16 + lm, DD, kb, g);
                v16bf_t b = frag_b_ld(St,      tn * 16 + lm, DD, kb, g);
                accO = WMMA_BF16(a, b, accO);
            }
            #pragma unroll
            for (int kb = 0; kb < CC; kb += TK) {
                v16bf_t a = frag_a_ld(Pg + cb, tm * 16 + lm, CC, kb, g);
                v16bf_t b = frag_b_ld(Ut,      tn * 16 + lm, CC, kb, g);
                accO = WMMA_BF16(a, b, accO);
            }
            #pragma unroll
            for (int r = 0; r < 8; ++r) {
                int row = tm * 16 + r + 8 * g;
                int col = tn * 16 + lm;
                obuf[((size_t)bh * LL + (size_t)c * CC + row) * DD + col] = accO[r];
            }

            v8f_t accS = {};
            #pragma unroll
            for (int kb = 0; kb < CC; kb += TK) {
                v16bf_t a = frag_a_ld(Kt, tm * 16 + lm, CC, kb, g);
                v16bf_t b = frag_b_ld(Ut, tn * 16 + lm, CC, kb, g);
                accS = WMMA_BF16(a, b, accS);
            }
            #pragma unroll
            for (int r = 0; r < 8; ++r) {
                int row = tm * 16 + r + 8 * g;             // dk
                int col = tn * 16 + lm;                    // dv
                Sf[row * DD + col] += accS[r];
            }
        }
        __syncthreads();

        // rebuild S^T bf16 for next chunk
        for (int i = tid; i < DD * DD; i += 256) {
            int dk = i >> 6, dv = i & 63;
            St[dv * DD + dk] = f2bf(Sf[i]);
        }
        __syncthreads();
    }
}

// ---------------------------------------------------------------------------
// Host-side orchestration
// ---------------------------------------------------------------------------
extern "C" void kernel_launch(void* const* d_in, const int* in_sizes, int n_in,
                              void* d_out, int out_size, void* d_ws, size_t ws_size,
                              hipStream_t stream) {
    const float* x  = (const float*)d_in[0];
    const float* Wq = (const float*)d_in[1];
    const float* Wk = (const float*)d_in[2];
    const float* Wv = (const float*)d_in[3];
    const float* Wb = (const float*)d_in[4];
    const float* Wo = (const float*)d_in[5];
    float* out = (float*)d_out;

    char* ws = (char*)d_ws;
    size_t off = 0;
    auto carve = [&](size_t bytes) -> char* {
        char* p = ws + off;
        off = (off + bytes + 255) & ~(size_t)255;
        return p;
    };
    unsigned short* xb  = (unsigned short*)carve((size_t)MROWS * DMODEL * 2);
    unsigned short* wqb = (unsigned short*)carve((size_t)DMODEL * HD * 2);
    unsigned short* wkb = (unsigned short*)carve((size_t)DMODEL * HD * 2);
    unsigned short* wvb = (unsigned short*)carve((size_t)DMODEL * HD * 2);
    unsigned short* wob = (unsigned short*)carve((size_t)HD * DMODEL * 2);
    float* qbuf = (float*)carve((size_t)BH * LL * DD * 4);
    float* kbuf = (float*)carve((size_t)BH * LL * DD * 4);
    float* vbuf = (float*)carve((size_t)BH * LL * DD * 4);
    float* bet  = (float*)carve((size_t)BH * LL * 4);
    float* obuf = (float*)carve((size_t)BH * LL * DD * 4);
    unsigned short* ob  = (unsigned short*)carve((size_t)MROWS * HD * 2);
    unsigned short* WgW = (unsigned short*)carve((size_t)BH * LL * DD * 2);
    unsigned short* Qg  = (unsigned short*)carve((size_t)BH * LL * DD * 2);
    unsigned short* Pg  = (unsigned short*)carve((size_t)BH * NC * CC * CC * 2);
    unsigned short* Ktg = (unsigned short*)carve((size_t)BH * LL * DD * 2);
    float*          Ubg = (float*)carve((size_t)BH * LL * DD * 4);
    (void)ws_size; (void)n_in; (void)in_sizes; (void)out_size;

    const int nx = MROWS * DMODEL;
    const int nw = DMODEL * HD;

    cvt_f32_bf16_kernel<<<(nx + 255) / 256, 256, 0, stream>>>(x, xb, nx);
    cvt_f32_bf16_kernel<<<(nw + 255) / 256, 256, 0, stream>>>(Wq, wqb, nw);
    cvt_f32_bf16_kernel<<<(nw + 255) / 256, 256, 0, stream>>>(Wk, wkb, nw);
    cvt_f32_bf16_kernel<<<(nw + 255) / 256, 256, 0, stream>>>(Wv, wvb, nw);
    cvt_f32_bf16_kernel<<<(nw + 255) / 256, 256, 0, stream>>>(Wo, wob, nw);

    dim3 ggrid(HD / TN, MROWS / TM);           // (8, 32)
    gemm_bf16_kernel<<<ggrid, 256, 0, stream>>>(xb, wqb, qbuf, 1);
    gemm_bf16_kernel<<<ggrid, 256, 0, stream>>>(xb, wkb, kbuf, 1);
    gemm_bf16_kernel<<<ggrid, 256, 0, stream>>>(xb, wvb, vbuf, 1);

    beta_kernel<<<(BB * LL * HH + 255) / 256, 256, 0, stream>>>(x, Wb, bet);

    delta_prep_kernel<<<BH * NC, 256, 0, stream>>>(qbuf, kbuf, vbuf, bet,
                                                   WgW, Qg, Pg, Ktg, Ubg);
    delta_seq_kernel<<<BH, 256, 0, stream>>>(WgW, Qg, Pg, Ktg, Ubg, obuf);

    o_to_bf16_kernel<<<(nx + 255) / 256, 256, 0, stream>>>(obuf, ob, nx);
    gemm_bf16_kernel<<<ggrid, 256, 0, stream>>>(ob, wob, out, 0);
}
